// DecoderRNN_72009421684948
// MI455X (gfx1250) — compile-verified, matrix operand order checked by reading, not run on hardware
//
#include <hip/hip_runtime.h>

#define BB 64
#define HH 512
#define LL 3
#define VV 4096
#define TE 512
#define TT 100

typedef __attribute__((ext_vector_type(16))) __bf16 v16bf;
typedef __attribute__((ext_vector_type(8)))  float  v8f;

union FragU { v16bf v; unsigned u[8]; };

// ---------------- bf16 helpers ----------------
__device__ inline unsigned short f2bf(float a) {
  unsigned ua = __float_as_uint(a);
  return (unsigned short)((ua + 0x7FFFu + ((ua >> 16) & 1u)) >> 16);  // RNE
}
__device__ inline unsigned pack_bf16(float a, float b) {
  return (unsigned)f2bf(a) | ((unsigned)f2bf(b) << 16);
}

// Stage one 64(rows) x 32(k) f32 tile into LDS as packed bf16 pairs (VALU path,
// used only for the one-shot Wa_states GEMM whose A is f32 `states`).
__device__ inline void stage_tile64x32_f32(unsigned* __restrict__ dst,
                                           const float* __restrict__ src,
                                           size_t ld, int rowBase, int kc, int tid) {
#pragma unroll
  for (int i = 0; i < 4; ++i) {
    int p  = tid * 4 + i;        // 0..1023
    int r  = p >> 4;             // 0..63
    int kp = (p & 15) * 2;       // 0,2,..,30
    const float* s = src + (size_t)(rowBase + r) * ld + (size_t)(kc + kp);
    dst[p] = pack_bf16(s[0], s[1]);
  }
}

// Async-DMA stage of a 64x32 bf16 tile (4KB): one GLOBAL_LOAD_ASYNC_TO_LDS_B128
// per thread (256 threads * 16B). Tracked by ASYNCcnt.
__device__ inline void async_stage_tile(unsigned* lds_tile,
                                        const unsigned short* __restrict__ src,
                                        size_t ld, int rowBase, int kc, int tid) {
  int row = tid >> 2;                  // 0..63
  int ch  = tid & 3;                   // 16B chunk within the 64B row
  const void* g = (const void*)(src + (size_t)(rowBase + row) * ld + (size_t)kc + ch * 8);
  unsigned ldsa = (unsigned)(unsigned long long)lds_tile + (unsigned)(tid * 16);
  asm volatile("global_load_async_to_lds_b128 %0, %1, off"
               :: "v"(ldsa), "v"(g)
               : "memory");
}
template <int N>
__device__ inline void wait_async_le() {
  asm volatile("s_wait_asynccnt %0" :: "i"(N) : "memory");
}

// CDNA5 16x16x32 bf16 A/B fragment gather from a 64x32 bf16 LDS tile.
// Lane l: row = row0 + (l&15); VGPR i holds K-pair kb = (i>>2)*16 + (l>>4)*8 + (i&3)*2.
__device__ inline v16bf load_frag(const unsigned* __restrict__ buf, int row0, int lane) {
  FragU f;
  int rm   = row0 + (lane & 15);
  int half = lane >> 4;
#pragma unroll
  for (int i = 0; i < 8; ++i)
    f.u[i] = buf[rm * 16 + (i >> 2) * 8 + half * 4 + (i & 3)];
  return f.v;
}

__device__ inline v8f wmma_bf16(v16bf a, v16bf b, v8f c) {
  return __builtin_amdgcn_wmma_f32_16x16x32_bf16(false, a, false, b, (short)0, c,
                                                 false, false);
}

__device__ inline float sigmoidf_(float x) { return 1.f / (1.f + __expf(-x)); }

// ---------------- GEMM (f32 inputs, convert-in-stage): C = A @ W^T + bias ----
// Used once for Wa_states. grid = (N/64, M/64), block = 256 (8 wave32).
__global__ __launch_bounds__(256) void gemm_bt_f32_kernel(
    const float* __restrict__ A, int lda,
    const float* __restrict__ W, int ldw,
    const float* __restrict__ bias,
    float* __restrict__ C, int ldc,
    int K, int act) {
  __shared__ unsigned As[64 * 16];
  __shared__ unsigned Ws[64 * 16];
  const int tid  = threadIdx.x;
  const int lane = tid & 31;
  const int wv   = tid >> 5;
  const int m0   = (wv & 3) * 16;
  const int n0   = (wv >> 2) * 32;
  const int rowBase = blockIdx.y * 64;
  const int colBase = blockIdx.x * 64;

  v8f acc[2] = {};
  for (int kc = 0; kc < K; kc += 32) {
    __syncthreads();
    stage_tile64x32_f32(As, A, (size_t)lda, rowBase, kc, tid);
    stage_tile64x32_f32(Ws, W, (size_t)ldw, colBase, kc, tid);
    __syncthreads();
    v16bf a = load_frag(As, m0, lane);
#pragma unroll
    for (int f = 0; f < 2; ++f) {
      v16bf b = load_frag(Ws, n0 + f * 16, lane);
      acc[f] = wmma_bf16(a, b, acc[f]);
    }
  }
  const int nIdx = lane & 15, half = lane >> 4;
#pragma unroll
  for (int f = 0; f < 2; ++f) {
    int col  = colBase + n0 + f * 16 + nIdx;
    float bv = bias ? bias[col] : 0.f;
#pragma unroll
    for (int r = 0; r < 8; ++r) {
      int row = rowBase + m0 + r + 8 * half;
      float v = acc[f][r] + bv;
      if (act == 1) v = fmaxf(v, 0.f);
      C[(size_t)row * ldc + col] = v;
    }
  }
}

// ---------------- GEMM (bf16 inputs, double-buffered async-DMA staging) ------
// A: M x K bf16 (lda), W: N x K bf16 (ldw). grid = (N/64, M/64), block = 256.
__global__ __launch_bounds__(256) void gemm_bt_bf16_kernel(
    const unsigned short* __restrict__ A, int lda,
    const unsigned short* __restrict__ W, int ldw,
    const float* __restrict__ bias,
    float* __restrict__ C, int ldc,
    int K, int act) {
  __shared__ unsigned As[2][64 * 16];
  __shared__ unsigned Ws[2][64 * 16];
  const int tid  = threadIdx.x;
  const int lane = tid & 31;
  const int wv   = tid >> 5;
  const int m0   = (wv & 3) * 16;
  const int n0   = (wv >> 2) * 32;
  const int rowBase = blockIdx.y * 64;
  const int colBase = blockIdx.x * 64;

  v8f acc[2] = {};
  const int nk = K / 32;
  // prologue: chunk 0 -> buffer 0
  async_stage_tile(As[0], A, (size_t)lda, rowBase, 0, tid);
  async_stage_tile(Ws[0], W, (size_t)ldw, colBase, 0, tid);
  for (int i = 0; i < nk; ++i) {
    const int cur = i & 1, nxt = cur ^ 1;
    if (i + 1 < nk) {
      // overlap: DMA chunk i+1 while we compute chunk i
      async_stage_tile(As[nxt], A, (size_t)lda, rowBase, (i + 1) * 32, tid);
      async_stage_tile(Ws[nxt], W, (size_t)ldw, colBase, (i + 1) * 32, tid);
      wait_async_le<2>();   // chunk i complete; chunk i+1 still in flight
    } else {
      wait_async_le<0>();
    }
    __syncthreads();        // all waves have chunk i resident
    v16bf a = load_frag(As[cur], m0, lane);
#pragma unroll
    for (int f = 0; f < 2; ++f) {
      v16bf b = load_frag(Ws[cur], n0 + f * 16, lane);
      acc[f] = wmma_bf16(a, b, acc[f]);
    }
    __syncthreads();        // protect buffer `cur` before it is re-issued
  }
  const int nIdx = lane & 15, half = lane >> 4;
#pragma unroll
  for (int f = 0; f < 2; ++f) {
    int col  = colBase + n0 + f * 16 + nIdx;
    float bv = bias ? bias[col] : 0.f;
#pragma unroll
    for (int r = 0; r < 8; ++r) {
      int row = rowBase + m0 + r + 8 * half;
      float v = acc[f][r] + bv;
      if (act == 1) v = fmaxf(v, 0.f);
      C[(size_t)row * ldc + col] = v;
    }
  }
}

// ---------------- fused GRU layer (bf16, double-buffered async staging) ------
// Xb: 64 x Kin bf16, Hprev: f32 + bf16 views, Wih/Whh: bf16 (1536 x K).
// grid = 8 (gate-column tiles of 64), block = 256.
__global__ __launch_bounds__(256) void gru_layer_kernel(
    const unsigned short* __restrict__ Xb, int Kin,
    const float* __restrict__ Hprev,
    const unsigned short* __restrict__ Hprev_bf,
    const unsigned short* __restrict__ Wih,
    const unsigned short* __restrict__ Whh,
    const float* __restrict__ bih,
    const float* __restrict__ bhh,
    float* __restrict__ Hnew,
    unsigned short* __restrict__ Hnew_bf) {
  __shared__ unsigned As[2][64 * 16];
  __shared__ unsigned Ws[2][3][64 * 16];
  const int tid  = threadIdx.x;
  const int lane = tid & 31;
  const int wv   = tid >> 5;
  const int m0   = (wv & 3) * 16;
  const int n0   = (wv >> 2) * 32;
  const int jb   = blockIdx.x * 64;   // gate-column base (0..511)

  v8f accI[3][2] = {};
  v8f accH[3][2] = {};

  // ---- pass 1: gi = X @ Wih^T ----
  {
    const int nk = Kin / 32;
    async_stage_tile(As[0], Xb, (size_t)Kin, 0, 0, tid);
#pragma unroll
    for (int g = 0; g < 3; ++g)
      async_stage_tile(Ws[0][g], Wih + (size_t)(g * HH + jb) * Kin, (size_t)Kin, 0, 0, tid);
    for (int i = 0; i < nk; ++i) {
      const int cur = i & 1, nxt = cur ^ 1;
      if (i + 1 < nk) {
        int kc = (i + 1) * 32;
        async_stage_tile(As[nxt], Xb, (size_t)Kin, 0, kc, tid);
#pragma unroll
        for (int g = 0; g < 3; ++g)
          async_stage_tile(Ws[nxt][g], Wih + (size_t)(g * HH + jb) * Kin, (size_t)Kin, 0, kc, tid);
        wait_async_le<4>();
      } else {
        wait_async_le<0>();
      }
      __syncthreads();
      v16bf a = load_frag(As[cur], m0, lane);
#pragma unroll
      for (int g = 0; g < 3; ++g)
#pragma unroll
        for (int f = 0; f < 2; ++f) {
          v16bf b = load_frag(Ws[cur][g], n0 + f * 16, lane);
          accI[g][f] = wmma_bf16(a, b, accI[g][f]);
        }
      __syncthreads();
    }
  }
  // ---- pass 2: gh = Hprev @ Whh^T ----
  {
    const int nk = HH / 32;
    async_stage_tile(As[0], Hprev_bf, (size_t)HH, 0, 0, tid);
#pragma unroll
    for (int g = 0; g < 3; ++g)
      async_stage_tile(Ws[0][g], Whh + (size_t)(g * HH + jb) * HH, (size_t)HH, 0, 0, tid);
    for (int i = 0; i < nk; ++i) {
      const int cur = i & 1, nxt = cur ^ 1;
      if (i + 1 < nk) {
        int kc = (i + 1) * 32;
        async_stage_tile(As[nxt], Hprev_bf, (size_t)HH, 0, kc, tid);
#pragma unroll
        for (int g = 0; g < 3; ++g)
          async_stage_tile(Ws[nxt][g], Whh + (size_t)(g * HH + jb) * HH, (size_t)HH, 0, kc, tid);
        wait_async_le<4>();
      } else {
        wait_async_le<0>();
      }
      __syncthreads();
      v16bf a = load_frag(As[cur], m0, lane);
#pragma unroll
      for (int g = 0; g < 3; ++g)
#pragma unroll
        for (int f = 0; f < 2; ++f) {
          v16bf b = load_frag(Ws[cur][g], n0 + f * 16, lane);
          accH[g][f] = wmma_bf16(a, b, accH[g][f]);
        }
      __syncthreads();
    }
  }
  // ---- GRU pointwise ----
  const int nIdx = lane & 15, half = lane >> 4;
#pragma unroll
  for (int f = 0; f < 2; ++f) {
    int j = jb + n0 + f * 16 + nIdx;             // gate column 0..511
    float bir = bih[j], biz = bih[HH + j], bin = bih[2 * HH + j];
    float bhr = bhh[j], bhz = bhh[HH + j], bhn = bhh[2 * HH + j];
#pragma unroll
    for (int r = 0; r < 8; ++r) {
      int row  = m0 + r + 8 * half;              // batch 0..63
      float hp = Hprev[(size_t)row * HH + j];
      float rg = sigmoidf_(accI[0][f][r] + bir + accH[0][f][r] + bhr);
      float zg = sigmoidf_(accI[1][f][r] + biz + accH[1][f][r] + bhz);
      float ng = tanhf(accI[2][f][r] + bin + rg * (accH[2][f][r] + bhn));
      float hv = (1.f - zg) * ng + zg * hp;
      Hnew[(size_t)row * HH + j] = hv;
      Hnew_bf[(size_t)row * HH + j] = f2bf(hv);
    }
  }
}

// ---------------- attention: scores + softmax + context ----------------
__global__ __launch_bounds__(256) void attn_kernel(
    const float* __restrict__ Wa,      // (B*TE) x 512
    const float* __restrict__ Uh,      // 64 x 512
    const float* __restrict__ vw,      // 512
    const float* __restrict__ vb,      // 1
    const int*   __restrict__ lengths, // 64
    const float* __restrict__ states,  // (B*TE) x 1024
    float* __restrict__ attns) {       // 64 x 1024
  const int b   = blockIdx.x;
  const int tid = threadIdx.x;
  const int lane = tid & 31, wv = tid >> 5;
  __shared__ float uh_s[HH], v_s[HH], e_s[TE], red_s[8];

  for (int i = tid; i < HH; i += 256) { uh_s[i] = Uh[b * HH + i]; v_s[i] = vw[i]; }
  __syncthreads();
  const int   len = lengths[b];
  const float vbv = vb[0];

  for (int rr = 0; rr < 64; ++rr) {
    int t = wv * 64 + rr;
    const float* wrow = Wa + (size_t)(b * TE + t) * HH;
    float a = 0.f;
    for (int j = lane; j < HH; j += 32)
      a += fmaxf(wrow[j] + uh_s[j], 0.f) * v_s[j];
    for (int o = 16; o; o >>= 1) a += __shfl_xor(a, o, 32);
    if (lane == 0) e_s[t] = (t >= len) ? -__builtin_inff() : (a + vbv);
  }
  __syncthreads();

  float pm = -__builtin_inff();
  for (int i = tid; i < TE; i += 256) pm = fmaxf(pm, e_s[i]);
  for (int o = 16; o; o >>= 1) pm = fmaxf(pm, __shfl_xor(pm, o, 32));
  if (lane == 0) red_s[wv] = pm;
  __syncthreads();
  float mx = red_s[0];
#pragma unroll
  for (int i = 1; i < 8; ++i) mx = fmaxf(mx, red_s[i]);
  __syncthreads();
  float ps = 0.f;
  for (int i = tid; i < TE; i += 256) {
    float ex = __expf(e_s[i] - mx);
    e_s[i] = ex;
    ps += ex;
  }
  for (int o = 16; o; o >>= 1) ps += __shfl_xor(ps, o, 32);
  if (lane == 0) red_s[wv] = ps;
  __syncthreads();
  float ssum = 0.f;
#pragma unroll
  for (int i = 0; i < 8; ++i) ssum += red_s[i];
  const float inv = 1.f / ssum;

  float a0 = 0.f, a1 = 0.f, a2 = 0.f, a3 = 0.f;
  for (int t = 0; t < TE; ++t) {
    float w = e_s[t] * inv;
    const float* srow = states + (size_t)(b * TE + t) * (2 * HH);
    a0 += w * srow[tid];
    a1 += w * srow[tid + 256];
    a2 += w * srow[tid + 512];
    a3 += w * srow[tid + 768];
  }
  attns[(size_t)b * (2 * HH) + tid]       = a0;
  attns[(size_t)b * (2 * HH) + tid + 256] = a1;
  attns[(size_t)b * (2 * HH) + tid + 512] = a2;
  attns[(size_t)b * (2 * HH) + tid + 768] = a3;
}

// ---------------- classifier + CE (fused, one block per (b,t)) ----------------
__global__ __launch_bounds__(256) void ce_kernel(
    const float* __restrict__ ys,    // (B*TT) x 512, row = b*TT + t
    const float* __restrict__ clsW,  // 4096 x 512
    const float* __restrict__ clsB,  // 4096
    const int*   __restrict__ gt,    // B x (TT+1)
    float* __restrict__ acc) {       // [0]=sum(ce*mask) [1]=sum(mask)
  const int bt  = blockIdx.x;
  const int b   = bt / TT, t = bt % TT;
  const int tid = threadIdx.x;
  const int lane = tid & 31, wv = tid >> 5;
  __shared__ float y_s[HH];
  __shared__ float lg_s[VV];
  __shared__ float red_s[8];

  for (int i = tid; i < HH; i += 256) y_s[i] = ys[(size_t)bt * HH + i];
  __syncthreads();

  for (int rr = 0; rr < 512; ++rr) {
    int vI = wv * 512 + rr;
    const float* wrow = clsW + (size_t)vI * HH;
    float a = 0.f;
    for (int j = lane; j < HH; j += 32) a += wrow[j] * y_s[j];
    for (int o = 16; o; o >>= 1) a += __shfl_xor(a, o, 32);
    if (lane == 0) lg_s[vI] = a + clsB[vI];
  }
  __syncthreads();

  float pm = -__builtin_inff();
  for (int i = tid; i < VV; i += 256) pm = fmaxf(pm, lg_s[i]);
  for (int o = 16; o; o >>= 1) pm = fmaxf(pm, __shfl_xor(pm, o, 32));
  if (lane == 0) red_s[wv] = pm;
  __syncthreads();
  float mx = red_s[0];
#pragma unroll
  for (int i = 1; i < 8; ++i) mx = fmaxf(mx, red_s[i]);
  __syncthreads();
  float ps = 0.f;
  for (int i = tid; i < VV; i += 256) ps += __expf(lg_s[i] - mx);
  for (int o = 16; o; o >>= 1) ps += __shfl_xor(ps, o, 32);
  if (lane == 0) red_s[wv] = ps;
  __syncthreads();
  if (tid == 0) {
    float ssum = 0.f;
#pragma unroll
    for (int i = 0; i < 8; ++i) ssum += red_s[i];
    int   tv  = gt[b * (TT + 1) + t + 1];
    float msk = (tv > 0) ? 1.f : 0.f;
    float lp  = lg_s[tv] - mx - __logf(ssum);
    atomicAdd(&acc[0], -lp * msk);
    atomicAdd(&acc[1], msk);
  }
}

// ---------------- small utility kernels ----------------
__global__ void cvt_bf16_kernel(const float* __restrict__ src,
                                unsigned short* __restrict__ dst, long long n2) {
  long long i = (long long)blockIdx.x * blockDim.x + threadIdx.x;   // pair index
  if (i < n2) {
    long long p = i * 2;
    *(unsigned*)(dst + p) = pack_bf16(src[p], src[p + 1]);
  }
}

__global__ void init_kernel(const float* __restrict__ init_h,
                            const float* __restrict__ init_y,
                            float* __restrict__ hA, unsigned short* __restrict__ hA_bf,
                            float* __restrict__ y0, float* __restrict__ acc) {
  int i = blockIdx.x * blockDim.x + threadIdx.x;
  if (i < 2) acc[i] = 0.f;
  if (i < LL * BB * HH) {
    int l = i / (BB * HH);
    int j = i % HH;
    float v = init_h[l * HH + j];
    hA[i] = v;
    hA_bf[i] = f2bf(v);
  }
  if (i < BB * HH) y0[i] = init_y[i % HH];
}

__global__ void embed_kernel(const float* __restrict__ embW,
                             const int* __restrict__ gt,
                             float* __restrict__ xs) {
  int bt = blockIdx.x;            // row = b*TT + t
  int b = bt / TT, t = bt % TT;
  int idx = gt[b * (TT + 1) + t];
  for (int i = threadIdx.x; i < HH; i += blockDim.x)
    xs[(size_t)bt * HH + i] = embW[(size_t)idx * HH + i];
}

// D[row] = bf16(concat(A[row], B[row])); one block per row (64 rows)
__global__ void concat2_bf16_kernel(const float* __restrict__ A, int lda, int wA,
                                    const float* __restrict__ Bm, int ldb, int wB,
                                    unsigned short* __restrict__ D) {
  int row = blockIdx.x;
  int w = wA + wB;
  for (int i = threadIdx.x; i < w; i += blockDim.x) {
    float v = (i < wA) ? A[(size_t)row * lda + i] : Bm[(size_t)row * ldb + (i - wA)];
    D[(size_t)row * w + i] = f2bf(v);
  }
}

__global__ void finalize_kernel(const float* __restrict__ acc, float* __restrict__ out) {
  out[0] = acc[0] / acc[1];
}

// ---------------- host ----------------
extern "C" void kernel_launch(void* const* d_in, const int* in_sizes, int n_in,
                              void* d_out, int out_size, void* d_ws, size_t ws_size,
                              hipStream_t stream) {
  const float* states  = (const float*)d_in[0];
  const int*   lengths = (const int*)d_in[1];
  const int*   gt      = (const int*)d_in[2];
  const float* embW    = (const float*)d_in[3];
  const float* init_h  = (const float*)d_in[4];
  const float* init_y  = (const float*)d_in[5];
  const float* Wih0    = (const float*)d_in[6];
  const float* WihR    = (const float*)d_in[7];
  const float* Whh     = (const float*)d_in[8];
  const float* bih     = (const float*)d_in[9];
  const float* bhh     = (const float*)d_in[10];
  const float* aWw     = (const float*)d_in[11];
  const float* aWb     = (const float*)d_in[12];
  const float* aUw     = (const float*)d_in[13];
  const float* aUb     = (const float*)d_in[14];
  const float* avw     = (const float*)d_in[15];
  const float* avb     = (const float*)d_in[16];
  const float* fcw     = (const float*)d_in[17];
  const float* fcb     = (const float*)d_in[18];
  const float* clsW    = (const float*)d_in[19];
  const float* clsB    = (const float*)d_in[20];
  float* out = (float*)d_out;

  // ---- workspace layout (bytes) ----
  char* wsb = (char*)d_ws;
  size_t off = 0;
  auto allocF = [&](size_t n) { float* p = (float*)(wsb + off); off += n * 4; return p; };
  auto allocH = [&](size_t n) { unsigned short* p = (unsigned short*)(wsb + off); off += n * 2;
                                off = (off + 3) & ~(size_t)3; return p; };
  float* xs    = allocF((size_t)BB * TT * HH);
  float* Wa    = allocF((size_t)BB * TE * HH);
  float* ys    = allocF((size_t)BB * TT * HH);
  float* hA    = allocF((size_t)LL * BB * HH);
  float* hB    = allocF((size_t)LL * BB * HH);
  float* y0    = allocF((size_t)BB * HH);
  float* Uh    = allocF((size_t)BB * HH);
  float* attns = allocF((size_t)BB * 2 * HH);
  float* accv  = allocF(2);
  unsigned short* Wih0b = allocH((size_t)3 * HH * 2 * HH);      // 1536 x 1024
  unsigned short* WihRb = allocH((size_t)2 * 3 * HH * HH);      // 2 x 1536 x 512
  unsigned short* Whhb  = allocH((size_t)LL * 3 * HH * HH);     // 3 x 1536 x 512
  unsigned short* aUwb  = allocH((size_t)HH * HH);
  unsigned short* fcwb  = allocH((size_t)HH * 3 * HH);
  unsigned short* cur0b = allocH((size_t)BB * 2 * HH);
  unsigned short* hAb   = allocH((size_t)LL * BB * HH);
  unsigned short* hBb   = allocH((size_t)LL * BB * HH);
  unsigned short* fcinb = allocH((size_t)BB * 3 * HH);

  const size_t LH = (size_t)BB * HH;        // per-layer h block
  const size_t WL = (size_t)3 * HH * HH;    // per-layer 1536x512 weight block

  // ---- one-time: weight conversion, init, embedding, Wa_states GEMM ----
  auto cvt = [&](const float* s, unsigned short* d, size_t n) {
    long long n2 = (long long)(n / 2);
    cvt_bf16_kernel<<<(unsigned)((n2 + 255) / 256), 256, 0, stream>>>(s, d, n2);
  };
  cvt(Wih0, Wih0b, (size_t)3 * HH * 2 * HH);
  cvt(WihR, WihRb, (size_t)2 * 3 * HH * HH);
  cvt(Whh,  Whhb,  (size_t)LL * 3 * HH * HH);
  cvt(aUw,  aUwb,  (size_t)HH * HH);
  cvt(fcw,  fcwb,  (size_t)HH * 3 * HH);

  init_kernel<<<(LL * BB * HH + 255) / 256, 256, 0, stream>>>(init_h, init_y, hA, hAb, y0, accv);
  embed_kernel<<<BB * TT, 256, 0, stream>>>(embW, gt, xs);

  // Wa = states @ attn_Ww^T + attn_Wb  (M=32768, N=512, K=1024), f32 path
  gemm_bt_f32_kernel<<<dim3(HH / 64, (BB * TE) / 64), 256, 0, stream>>>(
      states, 2 * HH, aWw, 2 * HH, aWb, Wa, HH, 2 * HH, 0);

  // ---- sequential decoder ----
  float* hIn = hA;  unsigned short* hInB = hAb;
  float* hOut = hB; unsigned short* hOutB = hBb;
  for (int t = 0; t < TT; ++t) {
    const float* yprev = (t == 0) ? y0 : (ys + (size_t)(t - 1) * HH);
    int ylda = (t == 0) ? HH : TT * HH;
    concat2_bf16_kernel<<<BB, 256, 0, stream>>>(yprev, ylda, HH,
                                                xs + (size_t)t * HH, TT * HH, HH, cur0b);
    gru_layer_kernel<<<8, 256, 0, stream>>>(cur0b, 2 * HH,
                                            hIn + 0 * LH, hInB + 0 * LH,
                                            Wih0b, Whhb + 0 * WL,
                                            bih + 0 * 3 * HH, bhh + 0 * 3 * HH,
                                            hOut + 0 * LH, hOutB + 0 * LH);
    gru_layer_kernel<<<8, 256, 0, stream>>>(hOutB + 0 * LH, HH,
                                            hIn + 1 * LH, hInB + 1 * LH,
                                            WihRb + 0 * WL, Whhb + 1 * WL,
                                            bih + 1 * 3 * HH, bhh + 1 * 3 * HH,
                                            hOut + 1 * LH, hOutB + 1 * LH);
    gru_layer_kernel<<<8, 256, 0, stream>>>(hOutB + 1 * LH, HH,
                                            hIn + 2 * LH, hInB + 2 * LH,
                                            WihRb + 1 * WL, Whhb + 2 * WL,
                                            bih + 2 * 3 * HH, bhh + 2 * 3 * HH,
                                            hOut + 2 * LH, hOutB + 2 * LH);
    // Uh = h2 @ attn_Uw^T + attn_Ub
    gemm_bt_bf16_kernel<<<dim3(HH / 64, 1), 256, 0, stream>>>(
        hOutB + 2 * LH, HH, aUwb, HH, aUb, Uh, HH, HH, 0);
    attn_kernel<<<BB, 256, 0, stream>>>(Wa, Uh, avw, avb, lengths, states, attns);
    // y = relu([attns, h2] @ fc_w^T + fc_b) -> ys[:, t, :]
    concat2_bf16_kernel<<<BB, 256, 0, stream>>>(attns, 2 * HH, 2 * HH,
                                                hOut + 2 * LH, HH, HH, fcinb);
    gemm_bt_bf16_kernel<<<dim3(HH / 64, 1), 256, 0, stream>>>(
        fcinb, 3 * HH, fcwb, 3 * HH, fcb, ys + (size_t)t * HH, TT * HH, 3 * HH, 1);

    float* tf = hIn; hIn = hOut; hOut = tf;
    unsigned short* tb = hInB; hInB = hOutB; hOutB = tb;
  }

  ce_kernel<<<BB * TT, 256, 0, stream>>>(ys, clsW, clsB, gt, accv);
  finalize_kernel<<<1, 1, 0, stream>>>(accv, out);

  (void)in_sizes; (void)n_in; (void)out_size; (void)ws_size;
}
